// Model_74749610819660
// MI455X (gfx1250) — matrix operand order repaired and checked by reading, not compiled, not run on hardware
//
#include <hip/hip_runtime.h>
#include <hip/hip_bf16.h>

// ---------------------------------------------------------------------------
// MoE (top-1) + mean-pool + head for MI455X (gfx1250, wave32, WMMA bf16)
// ---------------------------------------------------------------------------

typedef __bf16 bf16;
typedef __attribute__((ext_vector_type(16))) __bf16 bf16x16;
typedef __attribute__((ext_vector_type(8)))  __bf16 bf16x8;
typedef __attribute__((ext_vector_type(4)))  __bf16 bf16x4;
typedef __attribute__((ext_vector_type(8)))  float  v8f;

constexpr int Bn = 16;
constexpr int Sn = 2048;
constexpr int Dn = 1024;
constexpr int Hn = 4096;
constexpr int En = 4;
constexpr int Cn = 1000;
constexpr int Nn = Bn * Sn;          // 32768 tokens
constexpr int KC = 512;              // H-chunk width staged in LDS
constexpr int XP = 1032;             // LDS pitch for x tile (bank-conflict pad)
constexpr int HP = KC + 8;           // LDS pitch for h chunk
constexpr int NBUCK = 64;            // pooled-partial buckets (atomic spread)
constexpr int TILES_PER_E = Nn / 16; // 2048

// Workspace layout (bytes)
constexpr size_t XB_OFF    = 0;                                   // bf16 x   [N][D]
constexpr size_t W1T_OFF   = XB_OFF   + (size_t)Nn * Dn * 2;      // bf16 W1t [E][H][D]
constexpr size_t W2T_OFF   = W1T_OFF  + (size_t)En * Dn * Hn * 2; // bf16 W2t [E][D][H]
constexpr size_t PERM_OFF  = W2T_OFF  + (size_t)En * Hn * Dn * 2; // int  perm[E][N]
constexpr size_t GATE_OFF  = PERM_OFF + (size_t)En * Nn * 4;      // f32  gate[N]
constexpr size_t STATS_OFF = GATE_OFF + (size_t)Nn * 4;           // cnt[4] ints, sumP[4] at +64
constexpr size_t PPART_OFF = STATS_OFF + 256;                     // f32 [NBUCK][B][D]
constexpr size_t POOL_OFF  = PPART_OFF + (size_t)NBUCK * Bn * Dn * 4; // f32 [B][D]

static __device__ inline v8f wmma_bf16(bf16x16 a, bf16x16 b, v8f c) {
  return __builtin_amdgcn_wmma_f32_16x16x32_bf16(false, a, false, b, (short)0, c,
                                                 false, false);
}

// ---------------------------------------------------------------------------
__global__ void zero_kernel(float* __restrict__ p, int n) {
  int i = blockIdx.x * blockDim.x + threadIdx.x;
  if (i < n) p[i] = 0.0f;
}

// transpose [rows][cols] f32 -> [cols][rows] bf16, per blockIdx.z matrix
__global__ void transpose_cvt(const float* __restrict__ in, bf16* __restrict__ out,
                              int rows, int cols) {
  __shared__ float t[32][33];
  const size_t base = (size_t)blockIdx.z * rows * cols;
  const int c0 = blockIdx.x * 32, r0 = blockIdx.y * 32;
  for (int i = threadIdx.y; i < 32; i += 8)
    t[i][threadIdx.x] = in[base + (size_t)(r0 + i) * cols + c0 + threadIdx.x];
  __syncthreads();
  for (int i = threadIdx.y; i < 32; i += 8)
    out[base + (size_t)(c0 + i) * rows + r0 + threadIdx.x] = (bf16)t[threadIdx.x][i];
}

// ---------------------------------------------------------------------------
// One wave per token: gate logits, softmax/top-1, expert scatter, x -> bf16.
__global__ void gate_kernel(const float* __restrict__ x, const float* __restrict__ Wg,
                            const float* __restrict__ bg, bf16* __restrict__ xb,
                            int* __restrict__ perm, float* __restrict__ gate_out,
                            int* __restrict__ cnt, float* __restrict__ sumP) {
  __shared__ float sP[4];
  const int tid = threadIdx.x, wid = tid >> 5, lane = tid & 31;
  if (tid < 4) sP[tid] = 0.0f;
  __syncthreads();

  const int tok = blockIdx.x * 8 + wid;
  const float* xr = x + (size_t)tok * Dn;
  bf16* xo = xb + (size_t)tok * Dn;

  float a0 = 0.f, a1 = 0.f, a2 = 0.f, a3 = 0.f;
#pragma unroll
  for (int j = 0; j < 8; ++j) {
    const int d0 = j * 128 + lane * 4;
    const float4 v = *(const float4*)(xr + d0);
    const float* wg = Wg + (size_t)d0 * 4;
    const float4 w0 = *(const float4*)(wg);
    const float4 w1 = *(const float4*)(wg + 4);
    const float4 w2 = *(const float4*)(wg + 8);
    const float4 w3 = *(const float4*)(wg + 12);
    a0 += v.x * w0.x + v.y * w1.x + v.z * w2.x + v.w * w3.x;
    a1 += v.x * w0.y + v.y * w1.y + v.z * w2.y + v.w * w3.y;
    a2 += v.x * w0.z + v.y * w1.z + v.z * w2.z + v.w * w3.z;
    a3 += v.x * w0.w + v.y * w1.w + v.z * w2.w + v.w * w3.w;
    bf16x4 o;
    o[0] = (bf16)v.x; o[1] = (bf16)v.y; o[2] = (bf16)v.z; o[3] = (bf16)v.w;
    *(bf16x4*)(xo + d0) = o;
  }
#pragma unroll
  for (int m = 16; m > 0; m >>= 1) {
    a0 += __shfl_xor(a0, m, 32);
    a1 += __shfl_xor(a1, m, 32);
    a2 += __shfl_xor(a2, m, 32);
    a3 += __shfl_xor(a3, m, 32);
  }
  if (lane == 0) {
    float l[4] = {a0 + bg[0], a1 + bg[1], a2 + bg[2], a3 + bg[3]};
    int emax = 0;
    float m = l[0];
#pragma unroll
    for (int e = 1; e < 4; ++e)
      if (l[e] > m) { m = l[e]; emax = e; }
    float p[4], s = 0.f;
#pragma unroll
    for (int e = 0; e < 4; ++e) { p[e] = __expf(l[e] - m); s += p[e]; }
    const float inv = 1.0f / s;
    gate_out[tok] = p[emax] * inv;
    const int pos = atomicAdd(&cnt[emax], 1);
    perm[emax * Nn + pos] = tok;
#pragma unroll
    for (int e = 0; e < 4; ++e) atomicAdd(&sP[e], p[e] * inv);
  }
  __syncthreads();
  if (tid < 4) atomicAdd(&sumP[tid], sP[tid]);
}

__global__ void aux_kernel(const int* __restrict__ cnt, const float* __restrict__ sumP,
                           float* __restrict__ out) {
  if (threadIdx.x == 0 && blockIdx.x == 0) {
    const float invN = 1.0f / (float)Nn;
    float a = 0.f;
#pragma unroll
    for (int e = 0; e < 4; ++e) a += ((float)cnt[e] * invN) * (sumP[e] * invN);
    out[Bn * Cn] = (float)En * a;
  }
}

// ---------------------------------------------------------------------------
// Fused routed FFN: per block = one 16-token tile of one expert's list.
// 512 threads = 16 waves. Each wave owns 64 output columns (4 f32 C tiles).
__global__ __launch_bounds__(512) void ffn_kernel(
    const bf16* __restrict__ xb, const bf16* __restrict__ W1t,
    const bf16* __restrict__ W2t, const float* __restrict__ b1,
    const float* __restrict__ b2, const int* __restrict__ perm,
    const float* __restrict__ gate, const int* __restrict__ cnt,
    float* __restrict__ ppart) {
  const int e    = blockIdx.x >> 11;       // / TILES_PER_E (2048)
  const int tile = blockIdx.x & 2047;
  const int cnt_e = cnt[e];
  const int base  = tile * 16;
  if (base >= cnt_e) return;

  __shared__ bf16 sX[16 * XP];
  __shared__ bf16 sH[16 * HP];
  __shared__ int   sTok[16];
  __shared__ float sGate[16];

  const int tid = threadIdx.x, wid = tid >> 5, lane = tid & 31;
  const int colLane = lane & 15;          // N (or M) index within a 16-wide tile
  const int hiHalf  = (lane >> 4) & 1;    // lanes 16-31 handle the +offset half

  if (tid < 16) {
    const int r  = base + tid;
    const int tk = (r < cnt_e) ? perm[e * Nn + r] : perm[e * Nn + base];
    sTok[tid]  = tk;
    sGate[tid] = (r < cnt_e) ? gate[tk] : 0.0f;
  }
  __syncthreads();

  { // stage x tile: 16 rows x 1024 bf16 (each thread: one row, 4x16B chunks)
    const int row = tid >> 5;
    const bf16x8* src = (const bf16x8*)(xb + (size_t)sTok[row] * Dn);
    bf16x8* dst = (bf16x8*)(sX + row * XP);
#pragma unroll
    for (int j = 0; j < 4; ++j) {
      const int c8 = lane + j * 32;
      dst[c8] = src[c8];
    }
  }
  __syncthreads();

  v8f acc[4] = {};
  const int n0 = wid * 64;                 // this wave's output column base

  for (int hc = 0; hc < Hn; hc += KC) {
    // ---- Phase A: h[16][KC] = relu(x @ W1[:, hc:hc+KC] + b1), 2 tiles/wave
    const int c0 = hc + wid * 32;
#pragma unroll
    for (int ht = 0; ht < 2; ++ht) {
      const int colG = c0 + ht * 16 + colLane;                  // global h col
      const bf16* bp = W1t + ((size_t)(e * Hn + colG)) * Dn + hiHalf * 16;
      const bf16* arow = sX + colLane * XP;                     // A row = lane&15
      v8f c = {};
#pragma unroll 8
      for (int k = 0; k < Dn; k += 32) {
        bf16x8 alo = *(const bf16x8*)(arow + k + hiHalf * 16);
        bf16x8 ahi = *(const bf16x8*)(arow + k + 8 + hiHalf * 16);
        bf16x16 a = __builtin_shufflevector(alo, ahi, 0, 1, 2, 3, 4, 5, 6, 7, 8,
                                            9, 10, 11, 12, 13, 14, 15);
        bf16x16 b = *(const bf16x16*)(bp + k);
        c = wmma_bf16(a, b, c);
      }
      const float bv = b1[e * Hn + colG];
      const int cc = (c0 - hc) + ht * 16 + colLane;             // col in chunk
#pragma unroll
      for (int i = 0; i < 8; ++i) {
        float f = c[i] + bv;
        f = f > 0.0f ? f : 0.0f;
        const int row = i + hiHalf * 8;                          // C layout row
        sH[row * HP + cc] = (bf16)f;
      }
    }
    __syncthreads();

    // ---- Phase B: acc += h_chunk @ W2[hc:hc+KC, n0:n0+64]
    const bf16* hrow = sH + colLane * HP;
#pragma unroll 4
    for (int k2 = 0; k2 < KC; k2 += 32) {
      bf16x8 alo = *(const bf16x8*)(hrow + k2 + hiHalf * 16);
      bf16x8 ahi = *(const bf16x8*)(hrow + k2 + 8 + hiHalf * 16);
      bf16x16 a = __builtin_shufflevector(alo, ahi, 0, 1, 2, 3, 4, 5, 6, 7, 8, 9,
                                          10, 11, 12, 13, 14, 15);
      const int kG = hc + k2 + hiHalf * 16;
#pragma unroll
      for (int t = 0; t < 4; ++t) {
        const int col = n0 + t * 16 + colLane;
        bf16x16 b = *(const bf16x16*)(W2t + ((size_t)(e * Dn + col)) * Hn + kG);
        acc[t] = wmma_bf16(a, b, acc[t]);
      }
    }
    __syncthreads();
  }

  // ---- Epilogue: gate * (y + b2) accumulated into bucketed pooled partials
  float* pp = ppart + (size_t)(blockIdx.x & (NBUCK - 1)) * Bn * Dn;
#pragma unroll
  for (int t = 0; t < 4; ++t) {
    const int col = n0 + t * 16 + colLane;
    const float b2v = b2[e * Dn + col];
#pragma unroll
    for (int i = 0; i < 8; ++i) {
      const int row = i + hiHalf * 8;
      const float val = (acc[t][i] + b2v) * sGate[row];
      const int bb = sTok[row] >> 11;                           // token / S
      atomicAdd(pp + bb * Dn + col, val);
    }
  }
}

// ---------------------------------------------------------------------------
__global__ void reduce_pool(const float* __restrict__ part, float* __restrict__ pooled) {
  const int i = blockIdx.x * blockDim.x + threadIdx.x;  // B*D = 16384
  float s = 0.f;
#pragma unroll 8
  for (int k = 0; k < NBUCK; ++k) s += part[(size_t)k * Bn * Dn + i];
  pooled[i] = s * (1.0f / (float)Sn);
}

__global__ void head_kernel(const float* __restrict__ pooled,
                            const float* __restrict__ Wh,
                            const float* __restrict__ bh, float* __restrict__ out) {
  const int i = blockIdx.x * blockDim.x + threadIdx.x;
  if (i >= Bn * Cn) return;
  const int b = i / Cn, c = i % Cn;
  const float* pr = pooled + b * Dn;
  float s = bh[c];
#pragma unroll 4
  for (int d = 0; d < Dn; ++d) s += pr[d] * Wh[(size_t)d * Cn + c];
  out[i] = s;
}

// ---------------------------------------------------------------------------
extern "C" void kernel_launch(void* const* d_in, const int* in_sizes, int n_in,
                              void* d_out, int out_size, void* d_ws, size_t ws_size,
                              hipStream_t stream) {
  const float* x  = (const float*)d_in[0];
  const float* Wg = (const float*)d_in[1];
  const float* bg = (const float*)d_in[2];
  const float* W1 = (const float*)d_in[3];
  const float* b1 = (const float*)d_in[4];
  const float* W2 = (const float*)d_in[5];
  const float* b2 = (const float*)d_in[6];
  const float* Wh = (const float*)d_in[7];
  const float* bh = (const float*)d_in[8];
  float* out = (float*)d_out;

  char* w = (char*)d_ws;
  bf16*  xb     = (bf16*)(w + XB_OFF);
  bf16*  W1t    = (bf16*)(w + W1T_OFF);
  bf16*  W2t    = (bf16*)(w + W2T_OFF);
  int*   perm   = (int*)(w + PERM_OFF);
  float* gateA  = (float*)(w + GATE_OFF);
  int*   cnt    = (int*)(w + STATS_OFF);
  float* sumP   = (float*)(w + STATS_OFF + 64);
  float* ppart  = (float*)(w + PPART_OFF);
  float* pooled = (float*)(w + POOL_OFF);

  // 1) zero stats + pooled partial buckets + pooled
  {
    const int nz = (int)((256 + (size_t)NBUCK * Bn * Dn * 4 + Bn * Dn * 4) / 4);
    zero_kernel<<<(nz + 255) / 256, 256, 0, stream>>>((float*)(w + STATS_OFF), nz);
  }
  // 2) transpose+convert expert weights to bf16 (W1t:[E][H][D], W2t:[E][D][H])
  transpose_cvt<<<dim3(Hn / 32, Dn / 32, En), dim3(32, 8), 0, stream>>>(W1, W1t, Dn, Hn);
  transpose_cvt<<<dim3(Dn / 32, Hn / 32, En), dim3(32, 8), 0, stream>>>(W2, W2t, Hn, Dn);
  // 3) gating + top-1 routing + x->bf16
  gate_kernel<<<Nn / 8, 256, 0, stream>>>(x, Wg, bg, xb, perm, gateA, cnt, sumP);
  // 4) aux loss -> out[16000]
  aux_kernel<<<1, 64, 0, stream>>>(cnt, sumP, out);
  // 5) routed fused FFN (WMMA) -> bucketed pooled partials
  ffn_kernel<<<En * TILES_PER_E, 512, 0, stream>>>(xb, W1t, W2t, b1, b2, perm, gateA,
                                                   cnt, ppart);
  // 6) reduce buckets -> pooled (mean over S)
  reduce_pool<<<(Bn * Dn) / 256, 256, 0, stream>>>(ppart, pooled);
  // 7) head GEMM -> logits
  head_kernel<<<(Bn * Cn + 255) / 256, 256, 0, stream>>>(pooled, Wh, bh, out);
}
